// GCN_37452114821615
// MI455X (gfx1250) — matrix operand order repaired and checked by reading, not compiled, not run on hardware
//
#include <hip/hip_runtime.h>
#include <math.h>

// ---------------------------------------------------------------------------
// Types for CDNA5 WMMA (wave32, 16x16x32 f16 -> f32)
// ---------------------------------------------------------------------------
typedef _Float16 h16;
typedef __attribute__((ext_vector_type(16))) _Float16 v16h;
typedef __attribute__((ext_vector_type(8)))  _Float16 v8h;
typedef __attribute__((ext_vector_type(8)))  float    v8f;
typedef __attribute__((ext_vector_type(8)))  int      v8i;

#define NEGV (-1000000000.0f)

static __device__ __forceinline__ v8f wmma32(v16h a, v16h b, v8f c) {
  // D = A(16x32 f16) * B(32x16 f16) + C(16x16 f32)
  return __builtin_amdgcn_wmma_f32_16x16x32_f16(false, a, false, b,
                                                (short)0, c, false, false);
}

static __device__ __forceinline__ v16h mk16(v8h lo, v8h hi) {
  v16h r;
#pragma unroll
  for (int i = 0; i < 8; ++i) { r[i] = lo[i]; r[8 + i] = hi[i]; }
  return r;
}

// A-fragment: lane<16 holds row M=lane, K {k..k+7, k+16..k+23};
//             lane>=16 holds row M=lane-16, K {k+8..k+15, k+24..k+31}
static __device__ __forceinline__ v16h load_a(const h16* row, int lane, int k) {
  int lo = (lane < 16) ? k : k + 8;
  int hi = (lane < 16) ? k + 16 : k + 24;
  return mk16(*(const v8h*)(row + lo), *(const v8h*)(row + hi));
}
// B-fragment: lane holds column N=lane&15; lane<16 -> K {k..k+7, k+16..k+23}, lane>=16 -> K {k+8..k+15, k+24..k+31}
static __device__ __forceinline__ v16h load_b(const h16* row, int lane, int k) {
  int o = k + ((lane < 16) ? 0 : 8);
  return mk16(*(const v8h*)(row + o), *(const v8h*)(row + o + 16));
}

// ---------------------------------------------------------------------------
// f32 -> f16 convert
// ---------------------------------------------------------------------------
__global__ __launch_bounds__(256) void k_f32_to_f16(const float* __restrict__ in,
                                                    h16* __restrict__ out, int n) {
  for (int i = blockIdx.x * blockDim.x + threadIdx.x; i < n;
       i += gridDim.x * blockDim.x)
    out[i] = (h16)in[i];
}

// ---------------------------------------------------------------------------
// Linear: Y = X(8192x256,f16) @ W^T(256x256,f16 rows)   [one 16x16 tile / wave]
// mode 0: f16 row-major out   mode 1: f16 transposed (B,256,1024) out
// mode 2: f32 out + bias
// ---------------------------------------------------------------------------
__global__ __launch_bounds__(256) void k_linear(const h16* __restrict__ X,
                                                const h16* __restrict__ Wt,
                                                h16* __restrict__ Yh,
                                                float* __restrict__ Yf,
                                                const float* __restrict__ bias,
                                                int mode) {
  int lane = threadIdx.x & 31, wid = threadIdx.x >> 5;
  int gt = blockIdx.x * 8 + wid;          // 0..8191 tiles
  int tm = gt >> 4, tn = gt & 15;
  int row0 = tm * 16, n0 = tn * 16;
  const h16* arow = X + (size_t)(row0 + (lane & 15)) * 256;
  const h16* brow = Wt + (size_t)(n0 + (lane & 15)) * 256;
  v8f c = {};
#pragma unroll
  for (int k = 0; k < 256; k += 32)
    c = wmma32(load_a(arow, lane, k), load_b(brow, lane, k), c);

  int mb = (lane < 16) ? 0 : 8;
  int col = n0 + (lane & 15);
  if (mode == 0) {
#pragma unroll
    for (int r = 0; r < 8; ++r)
      Yh[(size_t)(row0 + mb + r) * 256 + col] = (h16)c[r];
  } else if (mode == 1) {
    int b = row0 >> 10;
    int il = (row0 & 1023) + mb;          // 8-aligned -> 16B aligned store
    v8h hv;
#pragma unroll
    for (int r = 0; r < 8; ++r) hv[r] = (h16)c[r];
    *(v8h*)(Yh + ((size_t)b * 256 + col) * 1024 + il) = hv;
  } else {
    float bv = bias[col];
#pragma unroll
    for (int r = 0; r < 8; ++r)
      Yf[(size_t)(row0 + mb + r) * 256 + col] = c[r] + bv;
  }
}

// ---------------------------------------------------------------------------
// Neighbor counts (row reductions over ner_mask); stores 1/clamp(nb,1)
// which: 0 = d rows, 1 = q rows
// ---------------------------------------------------------------------------
__global__ __launch_bounds__(256) void k_nb(const int* __restrict__ ner,
                                            const int* __restrict__ dm,
                                            const int* __restrict__ qm,
                                            float* __restrict__ inv_nb, int which) {
  __shared__ float red[256];
  int r = blockIdx.x; int b = r >> 10; int i = r & 1023;
  int rsel = which ? 1024 + i : i;
  const int* nrow = ner + ((size_t)b * 2048 + rsel) * 2048;
  float acc = 0.f;
  for (int j = threadIdx.x; j < 2048; j += 256) {
    int nv = nrow[j];
    if (nv) {
      if (j < 1024) {
        int c = dm[b * 1024 + j];
        if (which == 0 && j == i) c = 0;          // eye_d
        acc += (float)c;
      } else {
        int jq = j - 1024;
        int c = qm[b * 1024 + jq];
        if (which == 1 && jq == i) c = 0;         // eye_q
        acc += (float)c;
      }
    }
  }
  red[threadIdx.x] = acc;
  __syncthreads();
  for (int s = 128; s > 0; s >>= 1) {
    if (threadIdx.x < s) red[threadIdx.x] += red[threadIdx.x + s];
    __syncthreads();
  }
  if (threadIdx.x == 0) {
    int sm = which ? qm[r] : dm[r];
    float nb = sm ? red[0] : 0.f;
    if (nb < 1.f) nb = 1.f;
    inv_nb[r] = 1.f / nb;
  }
}

// ---------------------------------------------------------------------------
// Scores: S = scale * A @ Bt^T, masked (rowm*colm*ner) -> NEG
// ---------------------------------------------------------------------------
__global__ __launch_bounds__(256) void k_scores(const h16* __restrict__ A,
                                                const h16* __restrict__ Bt,
                                                const int* __restrict__ rowm,
                                                const int* __restrict__ colm,
                                                const int* __restrict__ ner,
                                                int row_off, int col_off,
                                                float scale, float* __restrict__ S) {
  int lane = threadIdx.x & 31, wid = threadIdx.x >> 5;
  int gt = blockIdx.x * 8 + wid;          // 8*64*64 tiles
  int b = gt >> 12; int rem = gt & 4095;
  int tm = rem >> 6, tn = rem & 63;
  int i0 = tm * 16, j0 = tn * 16;
  const h16* arow = A + ((size_t)b * 1024 + i0 + (lane & 15)) * 256;
  const h16* brow = Bt + ((size_t)b * 1024 + j0 + (lane & 15)) * 256;
  v8f c = {};
#pragma unroll
  for (int k = 0; k < 256; k += 32)
    c = wmma32(load_a(arow, lane, k), load_b(brow, lane, k), c);

  int mb = (lane < 16) ? 0 : 8;
  int j = j0 + (lane & 15);
  int cm = colm[b * 1024 + j];
#pragma unroll
  for (int r = 0; r < 8; ++r) {
    int i = i0 + mb + r;
    int m = rowm[b * 1024 + i] * cm *
            ner[((size_t)b * 2048 + row_off + i) * 2048 + col_off + j];
    S[((size_t)b * 1024 + i) * 1024 + j] = m ? c[r] * scale : NEGV;
  }
}

// ---------------------------------------------------------------------------
// Softmax over rows of 1024; masked entries (==NEG) stored as exact 0 in f16
// ---------------------------------------------------------------------------
__global__ __launch_bounds__(256) void k_softmax(const float* __restrict__ S,
                                                 h16* __restrict__ P) {
  __shared__ float red[256];
  size_t row = blockIdx.x;
  const float* s = S + row * 1024;
  float v[4], ex[4];
  float mx = -3.4e38f;
#pragma unroll
  for (int e = 0; e < 4; ++e) { v[e] = s[threadIdx.x + e * 256]; mx = fmaxf(mx, v[e]); }
  red[threadIdx.x] = mx; __syncthreads();
  for (int t = 128; t > 0; t >>= 1) {
    if (threadIdx.x < t) red[threadIdx.x] = fmaxf(red[threadIdx.x], red[threadIdx.x + t]);
    __syncthreads();
  }
  mx = red[0]; __syncthreads();
  float sum = 0.f;
#pragma unroll
  for (int e = 0; e < 4; ++e) { ex[e] = __expf(v[e] - mx); sum += ex[e]; }
  red[threadIdx.x] = sum; __syncthreads();
  for (int t = 128; t > 0; t >>= 1) {
    if (threadIdx.x < t) red[threadIdx.x] += red[threadIdx.x + t];
    __syncthreads();
  }
  float inv = 1.f / red[0];
  h16* p = P + row * 1024;
#pragma unroll
  for (int e = 0; e < 4; ++e)
    p[threadIdx.x + e * 256] = (h16)((v[e] == NEGV) ? 0.f : ex[e] * inv);
}

// ---------------------------------------------------------------------------
// Aggregation: out = relu(self + inv_nb * (Al@Vl + Ar@Vr + Bl@Wl + Br@Wr))
// A-fragments built on the fly: attn * graph (l) / attn * !graph (r) * !eye
// V buffers are pre-transposed (B,256,1024) so B-fragments are contiguous.
// ---------------------------------------------------------------------------
static __device__ __forceinline__ void agg_pair(const h16* __restrict__ attnrow,
                                                const int* __restrict__ grow,
                                                int diag, int ia, int j0, int lane,
                                                const h16* __restrict__ vl,
                                                const h16* __restrict__ vr, v8f& c) {
  int lo = (lane < 16) ? j0 : j0 + 8;
  int hi = (lane < 16) ? j0 + 16 : j0 + 24;
  v8h p0 = *(const v8h*)(attnrow + lo);
  v8h p1 = *(const v8h*)(attnrow + hi);
  v8i g0 = *(const v8i*)(grow + lo);
  v8i g1 = *(const v8i*)(grow + hi);
  v16h al, ar;
  h16 z = (h16)0.f;
#pragma unroll
  for (int e = 0; e < 8; ++e) {
    bool d0 = (!diag) || (lo + e != ia);
    bool d1 = (!diag) || (hi + e != ia);
    bool gl0 = g0[e] != 0, gl1 = g1[e] != 0;
    al[e]     = (d0 && gl0)  ? p0[e] : z;
    ar[e]     = (d0 && !gl0) ? p0[e] : z;
    al[8 + e] = (d1 && gl1)  ? p1[e] : z;
    ar[8 + e] = (d1 && !gl1) ? p1[e] : z;
  }
  int o = j0 + ((lane < 16) ? 0 : 8);
  v16h bl = mk16(*(const v8h*)(vl + o), *(const v8h*)(vl + o + 16));
  v16h br = mk16(*(const v8h*)(vr + o), *(const v8h*)(vr + o + 16));
  c = wmma32(al, bl, c);
  c = wmma32(ar, br, c);
}

__global__ __launch_bounds__(256) void k_agg(const h16* __restrict__ attnA,
                                             const h16* __restrict__ attnB,
                                             const int* __restrict__ graph,
                                             int grow_off, int gcolA, int gcolB, int diagA,
                                             const h16* __restrict__ VAl,
                                             const h16* __restrict__ VAr,
                                             const h16* __restrict__ VBl,
                                             const h16* __restrict__ VBr,
                                             const float* __restrict__ inv_nb,
                                             const float* __restrict__ selfb,
                                             float* __restrict__ out) {
  int lane = threadIdx.x & 31, wid = threadIdx.x >> 5;
  int gt = blockIdx.x * 8 + wid;          // 8*64*16 tiles
  int b = gt >> 10; int rem = gt & 1023;
  int tm = rem >> 4, tn = rem & 15;
  int i0 = tm * 16, n0 = tn * 16;
  int ia = i0 + (lane & 15);
  const h16* arowA = attnA + ((size_t)b * 1024 + ia) * 1024;
  const h16* arowB = attnB + ((size_t)b * 1024 + ia) * 1024;
  const int* grow = graph + ((size_t)b * 2048 + grow_off + ia) * 2048;
  size_t vb = ((size_t)b * 256 + n0 + (lane & 15)) * 1024;
  v8f c = {};
  for (int j0 = 0; j0 < 1024; j0 += 32) {
    agg_pair(arowA, grow + gcolA, diagA, ia, j0, lane, VAl + vb, VAr + vb, c);
    agg_pair(arowB, grow + gcolB, 0,     ia, j0, lane, VBl + vb, VBr + vb, c);
  }
  int mb = (lane < 16) ? 0 : 8;
  int col = n0 + (lane & 15);
#pragma unroll
  for (int r = 0; r < 8; ++r) {
    size_t ridx = (size_t)b * 1024 + i0 + mb + r;
    float val = c[r] * inv_nb[ridx] + selfb[ridx * 256 + col];
    out[ridx * 256 + col] = val > 0.f ? val : 0.f;
  }
}

// ---------------------------------------------------------------------------
// Host orchestration
// ---------------------------------------------------------------------------
extern "C" void kernel_launch(void* const* d_in, const int* in_sizes, int n_in,
                              void* d_out, int out_size, void* d_ws, size_t ws_size,
                              hipStream_t stream) {
  (void)in_sizes; (void)n_in; (void)out_size;
  const int NODE = 8 * 1024 * 256;                 // 2,097,152
  const size_t SC = (size_t)8 * 1024 * 1024;       // score elements

  // workspace carve-up
  size_t o = 0;
  auto take = [&](size_t bytes) -> size_t {
    size_t a = (o + 255) & ~(size_t)255; o = a + bytes; return a;
  };
  char* w = (char*)d_ws;
  size_t off_d16  = take((size_t)NODE * 2);
  size_t off_q16  = take((size_t)NODE * 2);
  size_t off_w16  = take((size_t)15 * 65536 * 2);
  size_t off_Qd   = take((size_t)NODE * 2);
  size_t off_Kd   = take((size_t)NODE * 2);
  size_t off_Qq   = take((size_t)NODE * 2);
  size_t off_Kq   = take((size_t)NODE * 2);
  size_t off_Fdq  = take((size_t)NODE * 2);
  size_t off_Fqd  = take((size_t)NODE * 2);
  size_t off_VT[8]; for (int i = 0; i < 8; ++i) off_VT[i] = take((size_t)NODE * 2);
  size_t off_selfd = take((size_t)NODE * 4);
  size_t off_selfq = take((size_t)NODE * 4);
  size_t off_invd  = take((size_t)8 * 1024 * 4);
  size_t off_invq  = take((size_t)8 * 1024 * 4);
  size_t off_S     = take(SC * 4);
  size_t off_attn[4]; for (int i = 0; i < 4; ++i) off_attn[i] = take(SC * 2);
  if (o > ws_size) return;   // workspace too small: bail deterministically

  h16*  d16 = (h16*)(w + off_d16);
  h16*  q16 = (h16*)(w + off_q16);
  h16*  w16 = (h16*)(w + off_w16);
  auto Wslot = [&](int s) { return w16 + (size_t)s * 65536; };
  h16 *Qd = (h16*)(w + off_Qd), *Kd = (h16*)(w + off_Kd);
  h16 *Qq = (h16*)(w + off_Qq), *Kq = (h16*)(w + off_Kq);
  h16 *Fdq = (h16*)(w + off_Fdq), *Fqd = (h16*)(w + off_Fqd);
  h16* VT[8]; for (int i = 0; i < 8; ++i) VT[i] = (h16*)(w + off_VT[i]);
  float* self_d = (float*)(w + off_selfd);
  float* self_q = (float*)(w + off_selfq);
  float* inv_d  = (float*)(w + off_invd);
  float* inv_q  = (float*)(w + off_invq);
  float* S      = (float*)(w + off_S);
  h16* attn_d  = (h16*)(w + off_attn[0]);
  h16* attn_q  = (h16*)(w + off_attn[1]);
  h16* attn_dq = (h16*)(w + off_attn[2]);
  h16* attn_qd = (h16*)(w + off_attn[3]);

  // inputs
  const float* d_node = (const float*)d_in[0];
  const float* q_node = (const float*)d_in[1];
  const float* bself  = (const float*)d_in[9];
  const int* dm   = (const int*)d_in[18];
  const int* qm   = (const int*)d_in[19];
  const int* ner  = (const int*)d_in[20];
  const int* gph  = (const int*)d_in[21];

  // w16 slot order -> d_in index
  // 0 W_dq_query 1 W_dk_key 2 W_qq_query 3 W_qk_key 4 W_dq_fc 5 W_qd_fc
  // 6 W_self 7 W_dd_l 8 W_qq_l 9 W_dq_l 10 W_qd_l 11 W_dd_r 12 W_qq_r 13 W_dq_r 14 W_qd_r
  const int widx[15] = {2, 3, 4, 5, 6, 7, 8, 10, 11, 12, 13, 14, 15, 16, 17};

  // 1) convert nodes + weights to f16
  k_f32_to_f16<<<512, 256, 0, stream>>>(d_node, d16, NODE);
  k_f32_to_f16<<<512, 256, 0, stream>>>(q_node, q16, NODE);
  for (int s = 0; s < 15; ++s)
    k_f32_to_f16<<<64, 256, 0, stream>>>((const float*)d_in[widx[s]], Wslot(s), 65536);

  // 2) linear projections (WMMA)
  k_linear<<<1024, 256, 0, stream>>>(d16, Wslot(0), Qd,  nullptr, nullptr, 0); // Qd
  k_linear<<<1024, 256, 0, stream>>>(d16, Wslot(1), Kd,  nullptr, nullptr, 0); // Kd
  k_linear<<<1024, 256, 0, stream>>>(q16, Wslot(2), Qq,  nullptr, nullptr, 0); // Qq
  k_linear<<<1024, 256, 0, stream>>>(q16, Wslot(3), Kq,  nullptr, nullptr, 0); // Kq
  k_linear<<<1024, 256, 0, stream>>>(q16, Wslot(4), Fdq, nullptr, nullptr, 0); // dq keys
  k_linear<<<1024, 256, 0, stream>>>(d16, Wslot(5), Fqd, nullptr, nullptr, 0); // qd keys
  // value projections stored transposed (B,256,1024)
  k_linear<<<1024, 256, 0, stream>>>(d16, Wslot(7),  VT[0], nullptr, nullptr, 1); // dd_l
  k_linear<<<1024, 256, 0, stream>>>(d16, Wslot(11), VT[1], nullptr, nullptr, 1); // dd_r
  k_linear<<<1024, 256, 0, stream>>>(q16, Wslot(9),  VT[2], nullptr, nullptr, 1); // dq_l
  k_linear<<<1024, 256, 0, stream>>>(q16, Wslot(13), VT[3], nullptr, nullptr, 1); // dq_r
  k_linear<<<1024, 256, 0, stream>>>(q16, Wslot(8),  VT[4], nullptr, nullptr, 1); // qq_l
  k_linear<<<1024, 256, 0, stream>>>(q16, Wslot(12), VT[5], nullptr, nullptr, 1); // qq_r
  k_linear<<<1024, 256, 0, stream>>>(d16, Wslot(10), VT[6], nullptr, nullptr, 1); // qd_l
  k_linear<<<1024, 256, 0, stream>>>(d16, Wslot(14), VT[7], nullptr, nullptr, 1); // qd_r
  // self projections (f32 + bias)
  k_linear<<<1024, 256, 0, stream>>>(d16, Wslot(6), nullptr, self_d, bself, 2);
  k_linear<<<1024, 256, 0, stream>>>(q16, Wslot(6), nullptr, self_q, bself, 2);

  // 3) neighbor counts
  k_nb<<<8192, 256, 0, stream>>>(ner, dm, qm, inv_d, 0);
  k_nb<<<8192, 256, 0, stream>>>(ner, dm, qm, inv_q, 1);

  // 4) attention: scores (WMMA) -> softmax (f16 attn, masked->0), S reused
  const float scl = 0.0625f; // 1/sqrt(256)
  k_scores<<<4096, 256, 0, stream>>>(Qd,  Kd,  dm, dm, ner,    0,    0, scl, S);
  k_softmax<<<8192, 256, 0, stream>>>(S, attn_d);
  k_scores<<<4096, 256, 0, stream>>>(Qq,  Kq,  qm, qm, ner, 1024, 1024, scl, S);
  k_softmax<<<8192, 256, 0, stream>>>(S, attn_q);
  k_scores<<<4096, 256, 0, stream>>>(d16, Fdq, dm, qm, ner,    0, 1024, 1.0f, S);
  k_softmax<<<8192, 256, 0, stream>>>(S, attn_dq);
  k_scores<<<4096, 256, 0, stream>>>(q16, Fqd, qm, dm, ner, 1024,    0, 1.0f, S);
  k_softmax<<<8192, 256, 0, stream>>>(S, attn_qd);

  // 5) aggregation (WMMA) + fused inv_nb / self / relu epilogue
  float* out_d = (float*)d_out;
  float* out_q = out_d + NODE;
  k_agg<<<1024, 256, 0, stream>>>(attn_d, attn_dq, gph, 0,    0, 1024, 1,
                                  VT[0], VT[1], VT[2], VT[3], inv_d, self_d, out_d);
  k_agg<<<1024, 256, 0, stream>>>(attn_q, attn_qd, gph, 1024, 1024, 0, 1,
                                  VT[4], VT[5], VT[6], VT[7], inv_q, self_q, out_q);
}
